// BitNetLinear_89489938580090
// MI455X (gfx1250) — 8-bit matrix operand order repaired and checked by reading, compiled, not run on hardware
//
#include <hip/hip_runtime.h>
#include <stdint.h>


typedef __attribute__((ext_vector_type(16))) int   v16i;
typedef __attribute__((ext_vector_type(8)))  float v8f;
typedef int v4i_t __attribute__((vector_size(16)));

#define M_TOT 8192
#define K_TOT 4096
#define N_TOT 16384

#define BM 128
#define BN 128
#define BK 128
#define LDS_STRIDE (BK + 16)
#define LDS_A_SZ (BM * LDS_STRIDE)
#define LDS_B_SZ (BN * LDS_STRIDE)
#define LDS_BUF  (LDS_A_SZ + LDS_B_SZ)

#define REDUCE_BLOCKS 1024

#define WS_ALPHA_OFF 0u
#define WS_PART_OFF  256u
#define WS_AQ_OFF    8192u
#define WS_WQ_OFF    (8192u + (size_t)M_TOT * (size_t)K_TOT)

#define FP8_P1 0x38u
#define FP8_M1 0xB8u

__device__ __forceinline__ void async_cp16(const uint8_t* g, uint8_t* l) {
#if __has_builtin(__builtin_amdgcn_global_load_async_to_lds_b128)
  __builtin_amdgcn_global_load_async_to_lds_b128(
      (__attribute__((address_space(1))) v4i_t*)g,
      (__attribute__((address_space(3))) v4i_t*)l, 0, 0);
#else
  asm volatile("global_load_async_to_lds_b128 %0, %1, off"
               :
               : "v"((uint32_t)(uintptr_t)l), "v"((uint64_t)(uintptr_t)g)
               : "memory");
#endif
}

__device__ __forceinline__ void wait_async0() {
#if __has_builtin(__builtin_amdgcn_s_wait_asynccnt)
  __builtin_amdgcn_s_wait_asynccnt(0);
#else
  asm volatile("s_wait_asynccnt 0x0" ::: "memory");
#endif
}

__global__ __launch_bounds__(256) void wsum_partial(const float4* __restrict__ w,
                                                    float* __restrict__ part, int n4) {
  __shared__ float sm[256];
  float s = 0.f;
  int stride = gridDim.x * blockDim.x;
  for (int i = blockIdx.x * blockDim.x + threadIdx.x; i < n4; i += stride) {
    float4 v = w[i];
    s += fabsf(v.x) + fabsf(v.y) + fabsf(v.z) + fabsf(v.w);
  }
  sm[threadIdx.x] = s;
  __syncthreads();
  for (int off = 128; off > 0; off >>= 1) {
    if ((int)threadIdx.x < off) sm[threadIdx.x] += sm[threadIdx.x + off];
    __syncthreads();
  }
  if (threadIdx.x == 0) part[blockIdx.x] = sm[0];
}

__global__ __launch_bounds__(256) void wsum_final(const float* __restrict__ part,
                                                  float* __restrict__ alpha) {
  __shared__ float sm[256];
  float s = 0.f;
  for (int i = threadIdx.x; i < REDUCE_BLOCKS; i += 256) s += part[i];
  sm[threadIdx.x] = s;
  __syncthreads();
  for (int off = 128; off > 0; off >>= 1) {
    if ((int)threadIdx.x < off) sm[threadIdx.x] += sm[threadIdx.x + off];
    __syncthreads();
  }
  if (threadIdx.x == 0)
    alpha[0] = sm[0] / (float)((long long)N_TOT * (long long)K_TOT);
}

__global__ __launch_bounds__(256) void quant_w(const float4* __restrict__ w,
                                               uint32_t* __restrict__ wq,
                                               const float* __restrict__ alpha_p) {
  int i = blockIdx.x * blockDim.x + threadIdx.x;
  float thr = 0.5f * alpha_p[0];
  float4 v = w[i];
  uint32_t b0 = (fabsf(v.x) > thr) ? (v.x > 0.f ? FP8_P1 : FP8_M1) : 0u;
  uint32_t b1 = (fabsf(v.y) > thr) ? (v.y > 0.f ? FP8_P1 : FP8_M1) : 0u;
  uint32_t b2 = (fabsf(v.z) > thr) ? (v.z > 0.f ? FP8_P1 : FP8_M1) : 0u;
  uint32_t b3 = (fabsf(v.w) > thr) ? (v.w > 0.f ? FP8_P1 : FP8_M1) : 0u;
  wq[i] = b0 | (b1 << 8) | (b2 << 16) | (b3 << 24);
}

__global__ __launch_bounds__(256) void quant_x(const float4* __restrict__ x,
                                               uint32_t* __restrict__ xq) {
  int i = blockIdx.x * blockDim.x + threadIdx.x;
  float4 v = x[i];
  uint32_t b0 = (v.x > 0.f) ? FP8_P1 : ((v.x < 0.f) ? FP8_M1 : 0u);
  uint32_t b1 = (v.y > 0.f) ? FP8_P1 : ((v.y < 0.f) ? FP8_M1 : 0u);
  uint32_t b2 = (v.z > 0.f) ? FP8_P1 : ((v.z < 0.f) ? FP8_M1 : 0u);
  uint32_t b3 = (v.w > 0.f) ? FP8_P1 : ((v.w < 0.f) ? FP8_M1 : 0u);
  xq[i] = b0 | (b1 << 8) | (b2 << 16) | (b3 << 24);
}

__global__ __launch_bounds__(256) void bitnet_gemm(const uint8_t* __restrict__ Aq,
                                                   const uint8_t* __restrict__ Wq,
                                                   const float* __restrict__ bias,
                                                   const float* __restrict__ alpha_p,
                                                   float* __restrict__ out) {
  __shared__ uint8_t lds[2 * LDS_BUF];

  const int tid  = threadIdx.x;
  const int lane = tid & 31;
  const int wave = tid >> 5;
  const int wm   = wave >> 1;
  const int wn   = wave & 1;
  const int bm = blockIdx.y * BM;
  const int bn = blockIdx.x * BN;

  const int lhalf  = lane >> 4;
  const int l16    = lane & 15;
  const int a_row0 = wm * 32 + l16;
  const int a_koff = lhalf * 16;
  const int b_koff = lhalf * 16;

  auto issue_tile = [&](int k0, int buf) {
    uint8_t* dA = lds + buf * LDS_BUF;
    uint8_t* dB = dA + LDS_A_SZ;
#pragma unroll
    for (int s = 0; s < 4; ++s) {
      int seg = tid * 4 + s;
      int row = seg >> 3;
      int col = (seg & 7) * 16;
      async_cp16(Aq + (size_t)(bm + row) * K_TOT + k0 + col,
                 dA + row * LDS_STRIDE + col);
      async_cp16(Wq + (size_t)(bn + row) * K_TOT + k0 + col,
                 dB + row * LDS_STRIDE + col);
    }
  };

  v8f acc[2][4] = {};

  issue_tile(0, 0);
  wait_async0();
  __syncthreads();

  const int KITERS = K_TOT / BK;
  for (int it = 0; it < KITERS; ++it) {
    const int cur = it & 1;
    if (it + 1 < KITERS) issue_tile((it + 1) * BK, cur ^ 1);

    const uint8_t* sA = lds + cur * LDS_BUF;
    const uint8_t* sB = sA + LDS_A_SZ;

    v16i a0, a1;
#pragma unroll
    for (int p = 0; p < 8; ++p) {
      uint2 d0 = *(const uint2*)(sA + a_row0 * LDS_STRIDE + (p >> 1) * 32 + (p & 1) * 8 + a_koff);
      uint2 d1 = *(const uint2*)(sA + (a_row0 + 16) * LDS_STRIDE + (p >> 1) * 32 + (p & 1) * 8 + a_koff);
      a0[2 * p] = (int)d0.x;  a0[2 * p + 1] = (int)d0.y;
      a1[2 * p] = (int)d1.x;  a1[2 * p + 1] = (int)d1.y;
    }

#pragma unroll
    for (int nsub = 0; nsub < 4; ++nsub) {
      const uint8_t* bp = sB + (wn * 64 + nsub * 16 + l16) * LDS_STRIDE + b_koff;
      v16i b;
#pragma unroll
      for (int q = 0; q < 4; ++q) {
        uint4 d = *(const uint4*)(bp + q * 32);
        b[4 * q]     = (int)d.x;
        b[4 * q + 1] = (int)d.y;
        b[4 * q + 2] = (int)d.z;
        b[4 * q + 3] = (int)d.w;
      }
      acc[0][nsub] = __builtin_amdgcn_wmma_f32_16x16x128_fp8_fp8(
          a0, b, (short)0, acc[0][nsub], false, false);
      acc[1][nsub] = __builtin_amdgcn_wmma_f32_16x16x128_fp8_fp8(
          a1, b, (short)0, acc[1][nsub], false, false);
    }

    wait_async0();
    __syncthreads();
  }

  const float alpha = alpha_p[0];
#pragma unroll
  for (int sub = 0; sub < 2; ++sub) {
    const int m_base = bm + wm * 32 + sub * 16 + lhalf * 8;
#pragma unroll
    for (int nsub = 0; nsub < 4; ++nsub) {
      int col = bn + wn * 64 + nsub * 16 + l16;
      float bv = bias[col];
#pragma unroll
      for (int r = 0; r < 8; ++r) {
        out[(size_t)(m_base + r) * N_TOT + col] = alpha * acc[sub][nsub][r] + bv;
      }
    }
  }
}

extern "C" void kernel_launch(void* const* d_in, const int* in_sizes, int n_in,
                              void* d_out, int out_size, void* d_ws, size_t ws_size,
                              hipStream_t stream) {
  const float* x    = (const float*)d_in[0];
  const float* w    = (const float*)d_in[1];
  const float* bias = (const float*)d_in[2];
  float* out = (float*)d_out;

  uint8_t* ws    = (uint8_t*)d_ws;
  float*   alpha = (float*)(ws + WS_ALPHA_OFF);
  float*   part  = (float*)(ws + WS_PART_OFF);
  uint8_t* Aq    = ws + WS_AQ_OFF;
  uint8_t* Wq    = ws + WS_WQ_OFF;

  const int wN4 = (N_TOT * K_TOT) / 4;
  const int xN4 = (M_TOT * K_TOT) / 4;

  wsum_partial<<<REDUCE_BLOCKS, 256, 0, stream>>>((const float4*)w, part, wN4);
  wsum_final<<<1, 256, 0, stream>>>(part, alpha);
  quant_w<<<wN4 / 256, 256, 0, stream>>>((const float4*)w, (uint32_t*)Wq, alpha);
  quant_x<<<xN4 / 256, 256, 0, stream>>>((const float4*)x, (uint32_t*)Aq);

  dim3 grid(N_TOT / BN, M_TOT / BM);
  bitnet_gemm<<<grid, 256, 0, stream>>>(Aq, Wq, bias, alpha, out);
}
